// GraphAttentionModule_1666447311043
// MI455X (gfx1250) — matrix operand order repaired and checked by reading, not compiled, not run on hardware
//
#include <hip/hip_runtime.h>

// ---------------------------------------------------------------------------
// GAT (2-layer) for MI455X / gfx1250: bf16 WMMA + async-LDS double buffering.
// N=4096, F_IN=512, F_HID=128, HEADS=4, F_OUT=64
// ---------------------------------------------------------------------------

#define ALPHA_SLOPE 0.2f
#define NEGINF -9.0e15f

typedef __bf16 bf16;
typedef bf16  v16bf __attribute__((ext_vector_type(16)));
typedef float v8f   __attribute__((ext_vector_type(8)));

#if defined(__HIP_DEVICE_COMPILE__) && \
    __has_builtin(__builtin_amdgcn_global_load_async_to_lds_b128) && \
    __has_builtin(__builtin_amdgcn_s_wait_asynccnt)
#define USE_ASYNC_LDS 1
typedef int v4i __attribute__((vector_size(16)));
typedef __attribute__((address_space(1))) v4i* gv4i_p;  // global-AS int4*
typedef __attribute__((address_space(3))) v4i* lv4i_p;  // LDS-AS int4*
#else
#define USE_ASYNC_LDS 0
#endif

// A-matrix (16x32, 16-bit) per-lane K index for element e (ISA 7.12.2):
//  lanes 0-15 (half=0): e0..7 -> K0..7,  e8..15 -> K16..23
//  lanes16-31 (half=1): e0..7 -> K8..15, e8..15 -> K24..31
__device__ __forceinline__ int kOfA(int half, int e) {
  return (e < 8) ? (8 * half + e) : (8 * half + e + 8);
}
// B-matrix (32x16, 16-bit): same K order as A (e0..7 -> K8h..8h+7, e8..15 -> K16+8h..).
__device__ __forceinline__ int kOfB(int half, int e) { return (e < 8) ? (8 * half + e) : (8 * half + e + 8); }

// ---------------------------------------------------------------------------
// Pack fp32 [M x K] row-major into WMMA A-fragment order.
// ---------------------------------------------------------------------------
__global__ void pack_a_bf16(const float* __restrict__ src, bf16* __restrict__ dst,
                            int M, int K) {
  const int lane = threadIdx.x & 31;
  const int wave = threadIdx.x >> 5;
  const int tile = blockIdx.x * (blockDim.x >> 5) + wave;
  const int ktiles = K >> 5;
  const int mt = tile / ktiles;
  const int kt = tile - mt * ktiles;
  const int half = lane >> 4;
  const float* s = src + (size_t)(mt * 16 + (lane & 15)) * K + kt * 32;
  bf16* d = dst + (size_t)tile * 512 + lane * 16;
#pragma unroll
  for (int e = 0; e < 16; ++e) d[e] = (bf16)s[kOfA(half, e)];
}

// ---------------------------------------------------------------------------
// Pack fp32 [K x Nc] row-major into WMMA B-fragment order.
// ---------------------------------------------------------------------------
__global__ void pack_b_bf16(const float* __restrict__ src, bf16* __restrict__ dst,
                            int K, int Nc, long srcZ, long dstZ) {
  const int lane = threadIdx.x & 31;
  const int wave = threadIdx.x >> 5;
  const int tile = blockIdx.x * (blockDim.x >> 5) + wave;
  const int ntiles = Nc >> 4;
  const int kt = tile / ntiles;
  const int nt = tile - kt * ntiles;
  const int half = lane >> 4;
  const float* s = src + (size_t)blockIdx.z * srcZ;
  bf16* d = dst + (size_t)blockIdx.z * dstZ + (size_t)tile * 512 + lane * 16;
  const int col = nt * 16 + (lane & 15);
#pragma unroll
  for (int e = 0; e < 16; ++e)
    d[e] = (bf16)s[(size_t)(kt * 32 + kOfB(half, e)) * Nc + col];
}

// ---------------------------------------------------------------------------
// C[M x Nc] = Apacked * Bpacked, f32 accumulate via WMMA bf16.
// ---------------------------------------------------------------------------
__global__ void gemm_bf16_wmma(const bf16* __restrict__ Ap, const bf16* __restrict__ Bp,
                               float* __restrict__ C, int M, int K, int Nc,
                               long aZ, long bZ, long cZ) {
  const int lane = threadIdx.x & 31;
  const int wave = threadIdx.x >> 5;
  const int mt = blockIdx.x;
  const int ct = blockIdx.y * (blockDim.x >> 5) + wave;
  const bf16* A = Ap + (size_t)blockIdx.z * aZ;
  const bf16* B = Bp + (size_t)blockIdx.z * bZ;
  float* Cz = C + (size_t)blockIdx.z * cZ;
  const int kchunks = K >> 5;
  const int ntiles = Nc >> 4;
  v8f acc = {0.f, 0.f, 0.f, 0.f, 0.f, 0.f, 0.f, 0.f};
  const bf16* aPtr = A + (size_t)mt * kchunks * 512 + lane * 16;
  const bf16* bPtr = B + (size_t)ct * 512 + lane * 16;
  for (int kc = 0; kc < kchunks; ++kc) {
    v16bf a = *(const v16bf*)(aPtr + (size_t)kc * 512);
    v16bf b = *(const v16bf*)(bPtr + (size_t)kc * ntiles * 512);
    acc = __builtin_amdgcn_wmma_f32_16x16x32_bf16(false, a, false, b,
                                                  (short)0, acc, false, false);
  }
  const int half = lane >> 4, n = lane & 15;
#pragma unroll
  for (int i = 0; i < 8; ++i)
    Cz[(size_t)(mt * 16 + i + 8 * half) * Nc + ct * 16 + n] = acc[i];
}

// ---------------------------------------------------------------------------
// Per-(head,row) dots: src[h][n] = h[h][n][:].aSrc[h][:], same for dst.
// ---------------------------------------------------------------------------
__global__ void dot_av(const float* __restrict__ hmat, const float* __restrict__ aSrc,
                       const float* __restrict__ aDst, float* __restrict__ outSrc,
                       float* __restrict__ outDst, int N, int F) {
  const int lane = threadIdx.x & 31;
  const int wave = threadIdx.x >> 5;
  const int gw = blockIdx.x * (blockDim.x >> 5) + wave;  // over H*N rows
  const int h = gw / N;
  const float* hr = hmat + (size_t)gw * F;
  const float* as = aSrc + (size_t)h * F;
  const float* ad = aDst + (size_t)h * F;
  float s = 0.f, d = 0.f;
  for (int o = lane; o < F; o += 32) {
    float v = hr[o];
    s += v * as[o];
    d += v * ad[o];
  }
#pragma unroll
  for (int off = 16; off > 0; off >>= 1) {
    s += __shfl_xor(s, off, 32);
    d += __shfl_xor(d, off, 32);
  }
  if (lane == 0) { outSrc[gw] = s; outDst[gw] = d; }
}

// ---------------------------------------------------------------------------
// Per-head max over dst coefficients (softmax stabilizer upper bound).
// ---------------------------------------------------------------------------
__global__ void head_max(const float* __restrict__ dstv, float* __restrict__ mxOut,
                         int N) {
  __shared__ float red[256];
  const float* d = dstv + (size_t)blockIdx.x * N;
  float m = NEGINF;
  for (int i = threadIdx.x; i < N; i += blockDim.x) m = fmaxf(m, d[i]);
  red[threadIdx.x] = m;
  __syncthreads();
  for (int s = 128; s > 0; s >>= 1) {
    if (threadIdx.x < s) red[threadIdx.x] = fmaxf(red[threadIdx.x], red[threadIdx.x + s]);
    __syncthreads();
  }
  if (threadIdx.x == 0) mxOut[blockIdx.x] = red[0];
}

// ---------------------------------------------------------------------------
// Fused masked-softmax attention + aggregation + ELU.
// One wave owns a 16-row tile x all F columns (T=F/16 accumulators).
// Stabilizer: mx_i = lrelu(src_i + max_j dst_j) >= true row max (lrelu is
// monotone; masking only lowers the max), so exp(e-mx) <= 1. Softmax is
// shift-invariant, so the result is exact.
// The V panel for the next 32-column chunk streams into LDS via async b128
// copies (double buffer), released by s_wait_asynccnt while the current chunk
// runs its T WMMAs.
// ---------------------------------------------------------------------------
template <int T>
__global__ void gat_attn_kernel(const int* __restrict__ adj,
                                const float* __restrict__ srcv,
                                const float* __restrict__ dstv,
                                const float* __restrict__ maxdst,
                                const bf16* __restrict__ Vp,  // [H][N/32][T][32][16]
                                float* __restrict__ out, int N, int outStride,
                                int headColStride) {
  const int lane = threadIdx.x & 31;
  const int wave = threadIdx.x >> 5;
  const int gw = blockIdx.x * (blockDim.x >> 5) + wave;
  const int rowTiles = N >> 4;
  const int h = gw / rowTiles;
  const int i0 = (gw - h * rowTiles) << 4;
  const int m = lane & 15;
  const int half = lane >> 4;
  const int row = i0 + m;

  const float src_m = srcv[(size_t)h * N + row];
  const float* dsth = dstv + (size_t)h * N;
  const size_t adjRow = (size_t)row * N;

  float mx = src_m + maxdst[h];
  mx = mx > 0.f ? mx : ALPHA_SLOPE * mx;

  v8f acc[T];
#pragma unroll
  for (int t = 0; t < T; ++t) acc[t] = (v8f){0.f, 0.f, 0.f, 0.f, 0.f, 0.f, 0.f, 0.f};
  float lsum = 0.f;

  const int kA0 = 8 * half;        // elements 0..7  -> columns jc+kA0..+7
  const int kA1 = 8 * half + 16;   // elements 8..15 -> columns jc+kA1..+7
  const bf16* VpH = Vp + (size_t)h * (N >> 5) * T * 512;

#if USE_ASYNC_LDS
  __shared__ __align__(16) bf16 vbuf[4][2][T * 512];
  // Async-copy one 32xF V panel (T*1024 bytes) as 2T wave-wide b128 transfers.
  auto issueChunk = [&](int chunk, int b) {
    char* g = (char*)const_cast<bf16*>(VpH + (size_t)chunk * T * 512) + lane * 16;
    char* l = (char*)(&vbuf[wave][b][0]) + lane * 16;
#pragma unroll
    for (int i = 0; i < 2 * T; ++i)
      __builtin_amdgcn_global_load_async_to_lds_b128(
          (gv4i_p)(g + i * 512), (lv4i_p)(l + i * 512), 0, 0);
  };
  issueChunk(0, 0);
#endif

  int buf = 0;
  for (int jc = 0; jc < N; jc += 32) {
#if USE_ASYNC_LDS
    // Prefetch the next panel into the other buffer (wraps harmlessly on the
    // last iteration), then wait until only those 2T copies remain pending:
    // async loads complete in order, so the current panel is resident.
    const int nextChunk = ((jc + 32) < N) ? ((jc >> 5) + 1) : 0;
    issueChunk(nextChunk, buf ^ 1);
    __builtin_amdgcn_s_wait_asynccnt(2 * T);
#endif

    alignas(16) int av[16];
    alignas(16) float dv[16];
    const int* ap = adj + adjRow + jc;
    const float* dp = dsth + jc;
    __builtin_prefetch(ap + 32, 0, 0);
    *(int4*)(av + 0)    = *(const int4*)(ap + kA0);
    *(int4*)(av + 4)    = *(const int4*)(ap + kA0 + 4);
    *(int4*)(av + 8)    = *(const int4*)(ap + kA1);
    *(int4*)(av + 12)   = *(const int4*)(ap + kA1 + 4);
    *(float4*)(dv + 0)  = *(const float4*)(dp + kA0);
    *(float4*)(dv + 4)  = *(const float4*)(dp + kA0 + 4);
    *(float4*)(dv + 8)  = *(const float4*)(dp + kA1);
    *(float4*)(dv + 12) = *(const float4*)(dp + kA1 + 4);

    v16bf A;
#pragma unroll
    for (int e = 0; e < 16; ++e) {
      float ev = src_m + dv[e];
      ev = ev > 0.f ? ev : ALPHA_SLOPE * ev;
      float pe = (av[e] > 0) ? __expf(ev - mx) : 0.f;
      lsum += pe;
      A[e] = (bf16)pe;
    }

#if USE_ASYNC_LDS
    const bf16* bbase = &vbuf[wave][buf][lane * 16];
#pragma unroll
    for (int t = 0; t < T; ++t) {
      v16bf B = *(const v16bf*)(bbase + (size_t)t * 512);
      acc[t] = __builtin_amdgcn_wmma_f32_16x16x32_bf16(false, A, false, B,
                                                       (short)0, acc[t], false, false);
    }
#else
    const bf16* bptr = VpH + (size_t)(jc >> 5) * T * 512 + lane * 16;
#pragma unroll
    for (int t = 0; t < T; ++t) {
      v16bf B = *(const v16bf*)(bptr + (size_t)t * 512);
      acc[t] = __builtin_amdgcn_wmma_f32_16x16x32_bf16(false, A, false, B,
                                                       (short)0, acc[t], false, false);
    }
#endif
    buf ^= 1;
  }
  (void)buf;

  // row sums: lanes l and l^16 each hold half of row m's sum
  lsum += __shfl_xor(lsum, 16, 32);
  __shared__ float sSum[4][16];
  if (half == 0) sSum[wave][m] = lsum;
  __syncthreads();

  // D layout: lane holds column n=lane&15, rows i + 8*half for i in 0..7
#pragma unroll
  for (int i = 0; i < 8; ++i) {
    const int r = i + 8 * half;
    const float inv = 1.f / sSum[wave][r];
#pragma unroll
    for (int t = 0; t < T; ++t) {
      float v = acc[t][i] * inv;
      v = v > 0.f ? v : expm1f(v);  // ELU (alpha=1)
      out[(size_t)(i0 + r) * outStride + h * headColStride + t * 16 + m] = v;
    }
  }
}

// ---------------------------------------------------------------------------
// Row-wise log_softmax over 64 columns; one wave per row.
// ---------------------------------------------------------------------------
__global__ void log_softmax64(const float* __restrict__ x, float* __restrict__ out) {
  const int lane = threadIdx.x & 31;
  const int wave = threadIdx.x >> 5;
  const int r = blockIdx.x * (blockDim.x >> 5) + wave;
  const float* xr = x + (size_t)r * 64;
  float v0 = xr[lane], v1 = xr[lane + 32];
  float mx = fmaxf(v0, v1);
#pragma unroll
  for (int off = 16; off > 0; off >>= 1) mx = fmaxf(mx, __shfl_xor(mx, off, 32));
  float s = __expf(v0 - mx) + __expf(v1 - mx);
#pragma unroll
  for (int off = 16; off > 0; off >>= 1) s += __shfl_xor(s, off, 32);
  const float lse = mx + __logf(s);
  out[(size_t)r * 64 + lane] = v0 - lse;
  out[(size_t)r * 64 + lane + 32] = v1 - lse;
}

// ---------------------------------------------------------------------------
// Host launcher
// ---------------------------------------------------------------------------
extern "C" void kernel_launch(void* const* d_in, const int* in_sizes, int n_in,
                              void* d_out, int out_size, void* d_ws, size_t ws_size,
                              hipStream_t stream) {
  const int N = 4096, FIN = 512, FH = 128, H = 4, FO = 64;

  const float* features = (const float*)d_in[0];
  const int*   adj      = (const int*)d_in[1];
  const float* W1       = (const float*)d_in[2];
  const float* a1_src   = (const float*)d_in[3];
  const float* a1_dst   = (const float*)d_in[4];
  const float* W2       = (const float*)d_in[5];
  const float* a2_src   = (const float*)d_in[6];
  const float* a2_dst   = (const float*)d_in[7];

  unsigned char* wsb = (unsigned char*)d_ws;
  size_t off = 0;
  auto alloc = [&](size_t bytes) -> void* {
    void* p = wsb + off;
    off += (bytes + 255) & ~(size_t)255;
    return p;
  };
  bf16*  xA   = (bf16*)alloc((size_t)N * FIN * 2);       // features, A-packed
  bf16*  W1B  = (bf16*)alloc((size_t)H * FIN * FH * 2);  // W1, B-packed
  bf16*  W2B  = (bf16*)alloc((size_t)FIN * FO * 2);      // W2, B-packed
  float* h1   = (float*)alloc((size_t)H * N * FH * 4);   // layer-1 features
  bf16*  h1B  = (bf16*)alloc((size_t)H * N * FH * 2);    // h1 as V, B-packed
  float* s1   = (float*)alloc((size_t)H * N * 4);
  float* t1   = (float*)alloc((size_t)H * N * 4);
  float* md1  = (float*)alloc((size_t)H * 4);
  float* x1   = (float*)alloc((size_t)N * 512 * 4);      // concat heads
  bf16*  x1A  = (bf16*)alloc((size_t)N * 512 * 2);       // x1, A-packed
  float* h2   = (float*)alloc((size_t)N * FO * 4);
  bf16*  h2B  = (bf16*)alloc((size_t)N * FO * 2);
  float* s2   = (float*)alloc((size_t)N * 4);
  float* t2   = (float*)alloc((size_t)N * 4);
  float* md2  = (float*)alloc((size_t)4);
  float* o2   = (float*)alloc((size_t)N * FO * 4);

  const dim3 blk(128);

  // ---- layer 1 ----------------------------------------------------------
  pack_a_bf16<<<1024, blk, 0, stream>>>(features, xA, N, FIN);
  pack_b_bf16<<<dim3(32, 1, H), blk, 0, stream>>>(W1, W1B, FIN, FH,
                                                  (long)FIN * FH, (long)FIN * FH);
  pack_b_bf16<<<dim3(16, 1, 1), blk, 0, stream>>>(W2, W2B, FIN, FO, 0, 0);

  gemm_bf16_wmma<<<dim3(N / 16, FH / 64, H), blk, 0, stream>>>(
      xA, W1B, h1, N, FIN, FH, 0L, (long)FIN * FH, (long)N * FH);

  dot_av<<<H * N / 4, blk, 0, stream>>>(h1, a1_src, a1_dst, s1, t1, N, FH);
  head_max<<<H, 256, 0, stream>>>(t1, md1, N);

  pack_b_bf16<<<dim3(256, 1, H), blk, 0, stream>>>(h1, h1B, N, FH,
                                                   (long)N * FH, (long)N * FH);

  gat_attn_kernel<8><<<H * (N / 16) / 4, blk, 0, stream>>>(adj, s1, t1, md1, h1B,
                                                           x1, N, 512, FH);

  // ---- layer 2 ----------------------------------------------------------
  pack_a_bf16<<<1024, blk, 0, stream>>>(x1, x1A, N, 512);
  gemm_bf16_wmma<<<dim3(N / 16, 1, 1), blk, 0, stream>>>(
      x1A, W2B, h2, N, 512, FO, 0L, 0L, 0L);
  dot_av<<<N / 4, blk, 0, stream>>>(h2, a2_src, a2_dst, s2, t2, N, FO);
  head_max<<<1, 256, 0, stream>>>(t2, md2, N);
  pack_b_bf16<<<dim3(128, 1, 1), blk, 0, stream>>>(h2, h2B, N, FO, 0, 0);
  gat_attn_kernel<4><<<(N / 16) / 4, blk, 0, stream>>>(adj, s2, t2, md2, h2B, o2,
                                                       N, FO, FO);

  // ---- output -----------------------------------------------------------
  log_softmax64<<<N / 4, blk, 0, stream>>>(o2, (float*)d_out);
}
